// EnhancedCrossAttention_68040871903543
// MI455X (gfx1250) — matrix operand order repaired and checked by reading, not compiled, not run on hardware
//
#include <hip/hip_runtime.h>
#include <hip/hip_bf16.h>

// ---------------------------------------------------------------------------
// EnhancedCrossAttention for MI455X (gfx1250): f16 WMMA pipeline, f32 accum,
// double-buffered LDS staging with async global->LDS copies where possible.
// ---------------------------------------------------------------------------

#define SREF 4096
#define DREF 512
#define HREF 8
#define DHREF 64

typedef _Float16 v16h __attribute__((ext_vector_type(16)));
typedef float    v8f  __attribute__((ext_vector_type(8)));
typedef unsigned int u32x4 __attribute__((ext_vector_type(4)));

union Frag { v16h v; u32x4 q[2]; };
union U4H  { u32x4 u; _Float16 h[8]; };
union F8   { v8f v; float f[8]; };

#if defined(__has_builtin)
#  if __has_builtin(__builtin_amdgcn_global_load_async_to_lds_b128)
#    define USE_ASYNC_LDS 1
#  endif
#endif
#ifndef USE_ASYNC_LDS
#  define USE_ASYNC_LDS 0
#endif

#if USE_ASYNC_LDS
typedef int v4i_async __attribute__((vector_size(16)));
typedef __attribute__((address_space(1))) v4i_async* as1_v4i;
typedef __attribute__((address_space(3))) v4i_async* as3_v4i;
#endif

// 16-byte global -> LDS copy (per-lane). Async on gfx1250 toolchains that
// expose the builtin (tracked on ASYNCcnt), else plain load+store.
__device__ __forceinline__ void copy16_g2l(const _Float16* g, _Float16* l) {
#if USE_ASYNC_LDS
  __builtin_amdgcn_global_load_async_to_lds_b128(
      (as1_v4i)(void*)g, (as3_v4i)(void*)l, 0, 0);
#else
  *(u32x4*)l = *(const u32x4*)g;
#endif
}

__device__ __forceinline__ void async_wait() {
#if USE_ASYNC_LDS
#  if __has_builtin(__builtin_amdgcn_s_wait_asynccnt)
  __builtin_amdgcn_s_wait_asynccnt(0);
#  else
  asm volatile("s_wait_asynccnt 0" ::: "memory");
#  endif
#endif
}

__device__ __forceinline__ v8f vzero8() {
  v8f z;
#pragma unroll
  for (int i = 0; i < 8; i++) z[i] = 0.0f;
  return z;
}

__device__ __forceinline__ v8f wmma_f16(const Frag& a, const Frag& b, v8f c) {
  return __builtin_amdgcn_wmma_f32_16x16x32_f16(false, a.v, false, b.v,
                                                (short)0, c, false, false);
}

// ------------------------------ elementwise --------------------------------

__global__ void cvt_f16_kernel(const float* __restrict__ x,
                               _Float16* __restrict__ y, int n) {
  int i = blockIdx.x * 256 + threadIdx.x;
  if (i < n) y[i] = (_Float16)x[i];
}

__global__ void concat_f16_kernel(const float* __restrict__ ctx,
                                  const float* __restrict__ emb,
                                  _Float16* __restrict__ out) {
  int i = blockIdx.x * 256 + threadIdx.x;   // over S*1024
  int s = i >> 10, j = i & 1023;
  float v = (j < DREF) ? ctx[s * DREF + j] : emb[s * DREF + (j - DREF)];
  out[i] = (_Float16)v;
}

__global__ void final_out_kernel(const float* __restrict__ emb,
                                 const float* __restrict__ gate,
                                 const float* __restrict__ proj,
                                 float* __restrict__ out, int n) {
  int i = blockIdx.x * 256 + threadIdx.x;
  if (i < n) out[i] = emb[i] + gate[i] * proj[i];
}

// ------------------------------ layernorms ---------------------------------

__global__ __launch_bounds__(256) void ln_f16_kernel(
    const float* __restrict__ X, const float* __restrict__ g,
    const float* __restrict__ b, _Float16* __restrict__ out) {
  int row = blockIdx.x, tid = threadIdx.x;
  const float* x = X + (size_t)row * DREF;
  float v0 = x[tid], v1 = x[tid + 256];
  float s = v0 + v1, ss = v0 * v0 + v1 * v1;
#pragma unroll
  for (int o = 16; o; o >>= 1) { s += __shfl_xor(s, o, 32); ss += __shfl_xor(ss, o, 32); }
  __shared__ float ws[8], wss[8];
  int w = tid >> 5, l = tid & 31;
  if (l == 0) { ws[w] = s; wss[w] = ss; }
  __syncthreads();
  float ts = 0.f, tss = 0.f;
#pragma unroll
  for (int i = 0; i < 8; i++) { ts += ws[i]; tss += wss[i]; }
  float mu = ts * (1.0f / DREF);
  float var = tss * (1.0f / DREF) - mu * mu;
  float rs = rsqrtf(var + 1e-5f);
  out[(size_t)row * DREF + tid]       = (_Float16)((v0 - mu) * rs * g[tid] + b[tid]);
  out[(size_t)row * DREF + tid + 256] = (_Float16)((v1 - mu) * rs * g[tid + 256] + b[tid + 256]);
}

__global__ __launch_bounds__(256) void gate_ln_sigmoid_kernel(
    float* __restrict__ X, const float* __restrict__ g,
    const float* __restrict__ b) {
  int row = blockIdx.x, tid = threadIdx.x;
  float* x = X + (size_t)row * DREF;
  float v0 = x[tid], v1 = x[tid + 256];
  float s = v0 + v1, ss = v0 * v0 + v1 * v1;
#pragma unroll
  for (int o = 16; o; o >>= 1) { s += __shfl_xor(s, o, 32); ss += __shfl_xor(ss, o, 32); }
  __shared__ float ws[8], wss[8];
  int w = tid >> 5, l = tid & 31;
  if (l == 0) { ws[w] = s; wss[w] = ss; }
  __syncthreads();
  float ts = 0.f, tss = 0.f;
#pragma unroll
  for (int i = 0; i < 8; i++) { ts += ws[i]; tss += wss[i]; }
  float mu = ts * (1.0f / DREF);
  float var = tss * (1.0f / DREF) - mu * mu;
  float rs = rsqrtf(var + 1e-5f);
  float z0 = (v0 - mu) * rs * g[tid] + b[tid];
  float z1 = (v1 - mu) * rs * g[tid + 256] + b[tid + 256];
  x[tid]       = 1.0f / (1.0f + __builtin_amdgcn_exp2f(-z0 * 1.44269504f));
  x[tid + 256] = 1.0f / (1.0f + __builtin_amdgcn_exp2f(-z1 * 1.44269504f));
}

// Split [S,512] f32 into per-head [H][S][64] f16, optional row-l2norm*scale.
__global__ __launch_bounds__(256) void split_heads_kernel(
    const float* __restrict__ X, _Float16* __restrict__ out,
    int S, float scale, int do_norm) {
  int w = threadIdx.x >> 5, l = threadIdx.x & 31;
  int p = blockIdx.x * 8 + w;           // in [0, S*H)
  int s = p >> 3, h = p & 7;
  const float* x = X + (size_t)s * DREF + h * DHREF;
  float e0 = x[l], e1 = x[l + 32];
  float inv = scale;
  if (do_norm) {
    float ss = e0 * e0 + e1 * e1;
#pragma unroll
    for (int o = 16; o; o >>= 1) ss += __shfl_xor(ss, o, 32);
    inv = scale / fmaxf(sqrtf(ss), 1e-12f);
  }
  _Float16* o = out + ((size_t)h * S + s) * DHREF;
  o[l]      = (_Float16)(e0 * inv);
  o[l + 32] = (_Float16)(e1 * inv);
}

// ------------------------------ WMMA GEMM ----------------------------------
// C[M,N] = A[M,K](f16,row) @ B[K,N](f16,row) + bias.  M%128==0,N%64==0,K%32==0.
// 256 threads = 8 waves; block tile 128x64; wave tile 16x64; K-step 32;
// ping-pong LDS double buffering, A tile staged via async global->LDS.
__global__ __launch_bounds__(256) void gemm_wmma_kernel(
    const _Float16* __restrict__ A, const _Float16* __restrict__ B,
    const float* __restrict__ bias, float* __restrict__ C,
    int M, int N, int K) {
  __shared__ alignas(16) _Float16 ldsA[2][128 * 32];   // [row][k]   2x8 KB
  __shared__ alignas(16) _Float16 ldsBt[2][64 * 32];   // [col][k]   2x4 KB
  int tid = threadIdx.x, w = tid >> 5, l = tid & 31;
  int m0 = blockIdx.x * 128, n0 = blockIdx.y * 64;
  int lq = l & 15, kb = (l >> 4) * 8, kh16 = (l >> 4) * 16;
  int arow = w * 16 + lq;
  int bk = tid >> 3, bgrp = (tid & 7) * 8;

  v8f acc[4];
#pragma unroll
  for (int j = 0; j < 4; j++) acc[j] = vzero8();

  auto stage = [&](int k0, int p) {
#pragma unroll
    for (int i = 0; i < 2; i++) {          // A tile 128x32: 512 16B copies
      int idx = tid + i * 256;
      int r = idx >> 2, q = (idx & 3) * 8;
      copy16_g2l(&A[(size_t)(m0 + r) * K + k0 + q], &ldsA[p][r * 32 + q]);
    }
    U4H u;                                  // B tile 32x64, transposed
    u.u = *(const u32x4*)&B[(size_t)(k0 + bk) * N + n0 + bgrp];
#pragma unroll
    for (int c = 0; c < 8; c++) ldsBt[p][(bgrp + c) * 32 + bk] = u.h[c];
  };

  stage(0, 0);
  async_wait();
  __syncthreads();

  int p = 0;
  for (int k0 = 0; k0 < K; k0 += 32) {
    if (k0 + 32 < K) stage(k0 + 32, p ^ 1);
    if (k0 + 64 < K)
      __builtin_prefetch(&A[(size_t)(m0 + arow) * K + k0 + 64], 0, 1);

    Frag af;
    af.q[0] = *(const u32x4*)&ldsA[p][arow * 32 + kb];
    af.q[1] = *(const u32x4*)&ldsA[p][arow * 32 + 16 + kb];
    Frag bf[4];
#pragma unroll
    for (int j = 0; j < 4; j++) {
      int col = j * 16 + lq;
      bf[j].q[0] = *(const u32x4*)&ldsBt[p][col * 32 + kb];
      bf[j].q[1] = *(const u32x4*)&ldsBt[p][col * 32 + 16 + kb];
    }
#pragma unroll
    for (int j = 0; j < 4; j++) acc[j] = wmma_f16(af, bf[j], acc[j]);

    async_wait();
    __syncthreads();
    p ^= 1;
  }

#pragma unroll
  for (int j = 0; j < 4; j++) {
    int col = n0 + j * 16 + lq;
    float bv = bias ? bias[col] : 0.0f;
    F8 t; t.v = acc[j];
#pragma unroll
    for (int v = 0; v < 8; v++) {
      int row = m0 + w * 16 + v + ((l >> 4) << 3);
      C[(size_t)row * N + col] = t.f[v] + bv;
    }
  }
}

// ---------------------------- WMMA attention -------------------------------
// qh/kh/vh: [H][S][64] f16 (q pre-scaled & l2-normed; k l2-normed).
// ctx: [Sq][512] f32. Grid: (Sq/128, H); 8 waves x 16-query tiles per block.
// Scores computed transposed (K_tile x Q^T) so the exp'd P lands directly in
// A-fragment layout. Clip to +/-10 => fixed-shift softmax exp(s-10) is exact.
__global__ __launch_bounds__(256) void attn_wmma_kernel(
    const _Float16* __restrict__ qh, const _Float16* __restrict__ kh,
    const _Float16* __restrict__ vh, float* __restrict__ ctx,
    int Sq, int Sk) {
  __shared__ alignas(16) _Float16 ldsK[2][32 * 64];    // [key][dh]  2x4 KB
  __shared__ alignas(16) _Float16 ldsVt[2][64 * 32];   // [dh][key]  2x4 KB
  int tid = threadIdx.x, w = tid >> 5, l = tid & 31;
  int head = blockIdx.y;
  int q0 = blockIdx.x * 128 + w * 16;
  int lq = l & 15, kb = (l >> 4) * 8, kh16 = (l >> 4) * 16;
  int sk = tid >> 3, sgrp = (tid & 7) * 8;

  // Q^T B-fragments for dh chunks 0-31 / 32-63 (kept in VGPRs all kernel)
  Frag qf[2];
  const _Float16* qrowp = qh + ((size_t)head * Sq + (q0 + lq)) * DHREF;
#pragma unroll
  for (int c = 0; c < 2; c++) {
    int base = c * 32 + kb;
    qf[c].q[0] = *(const u32x4*)&qrowp[base];
    qf[c].q[1] = *(const u32x4*)&qrowp[base + 16];
  }
  Frag ones;
#pragma unroll
  for (int i = 0; i < 16; i++) ones.v[i] = (_Float16)1.0f;

  v8f cacc[4];
#pragma unroll
  for (int j = 0; j < 4; j++) cacc[j] = vzero8();
  v8f sacc = vzero8();

  auto stage = [&](int kt, int p) {
    size_t gro = ((size_t)head * Sk + kt + sk) * DHREF + sgrp;
    copy16_g2l(&kh[gro], &ldsK[p][sk * 64 + sgrp]);   // K tile (row-major)
    U4H u; u.u = *(const u32x4*)&vh[gro];             // V tile (transposed)
#pragma unroll
    for (int c = 0; c < 8; c++) ldsVt[p][(sgrp + c) * 32 + sk] = u.h[c];
  };

  stage(0, 0);
  async_wait();
  __syncthreads();

  int p = 0;
  for (int kt = 0; kt < Sk; kt += 32) {
    if (kt + 32 < Sk) stage(kt + 32, p ^ 1);

    // S^T tiles: rows = keys, cols = queries
    v8f st0 = vzero8(), st1 = vzero8();
#pragma unroll
    for (int c = 0; c < 2; c++) {
      Frag ka, kbf;
      int o0 = lq * 64 + c * 32 + kb;
      ka.q[0] = *(const u32x4*)&ldsK[p][o0];
      ka.q[1] = *(const u32x4*)&ldsK[p][o0 + 16];
      st0 = wmma_f16(ka, qf[c], st0);
      int o1 = (16 + lq) * 64 + c * 32 + kb;
      kbf.q[0] = *(const u32x4*)&ldsK[p][o1];
      kbf.q[1] = *(const u32x4*)&ldsK[p][o1 + 16];
      st1 = wmma_f16(kbf, qf[c], st1);
    }

    // clip, fixed-shift exp; pack straight into A-fragment layout for P
    Frag pf;
    F8 a0, a1; a0.v = st0; a1.v = st1;
#pragma unroll
    for (int i = 0; i < 8; i++) {
      float s0 = fminf(fmaxf(a0.f[i], -10.f), 10.f);
      float s1 = fminf(fmaxf(a1.f[i], -10.f), 10.f);
      pf.v[i]     = (_Float16)__builtin_amdgcn_exp2f((s0 - 10.f) * 1.44269504f);
      pf.v[i + 8] = (_Float16)__builtin_amdgcn_exp2f((s1 - 10.f) * 1.44269504f);
    }

    // preload all V fragments, then issue the 5 WMMAs back-to-back
    Frag vf[4];
#pragma unroll
    for (int j = 0; j < 4; j++) {
      int col = j * 16 + lq;
      vf[j].q[0] = *(const u32x4*)&ldsVt[p][col * 32 + kb];
      vf[j].q[1] = *(const u32x4*)&ldsVt[p][col * 32 + 16 + kb];
    }
    sacc = wmma_f16(pf, ones, sacc);          // denominator accumulation
#pragma unroll
    for (int j = 0; j < 4; j++) cacc[j] = wmma_f16(pf, vf[j], cacc[j]);

    async_wait();
    __syncthreads();
    p ^= 1;
  }

  F8 sv; sv.v = sacc;
#pragma unroll
  for (int j = 0; j < 4; j++) {
    F8 t; t.v = cacc[j];
#pragma unroll
    for (int v = 0; v < 8; v++) {
      int row = q0 + v + ((l >> 4) << 3);
      ctx[(size_t)row * DREF + head * DHREF + j * 16 + lq] =
          t.f[v] / fmaxf(sv.f[v], 1e-20f);
    }
  }
}

// ------------------------------- launcher ----------------------------------

extern "C" void kernel_launch(void* const* d_in, const int* in_sizes, int n_in,
                              void* d_out, int out_size, void* d_ws, size_t ws_size,
                              hipStream_t stream) {
  const int S = SREF, D = DREF, H = HREF;
  const float* gene  = (const float*)d_in[0];
  const float* drug  = (const float*)d_in[1];
  const float* lng_g = (const float*)d_in[2];
  const float* lng_b = (const float*)d_in[3];
  const float* lnd_g = (const float*)d_in[4];
  const float* lnd_b = (const float*)d_in[5];
  const float* wgq = (const float*)d_in[6];  const float* bgq = (const float*)d_in[7];
  const float* wgk = (const float*)d_in[8];  const float* bgk = (const float*)d_in[9];
  const float* wgv = (const float*)d_in[10]; const float* bgv = (const float*)d_in[11];
  const float* wdq = (const float*)d_in[12]; const float* bdq = (const float*)d_in[13];
  const float* wdk = (const float*)d_in[14]; const float* bdk = (const float*)d_in[15];
  const float* wdv = (const float*)d_in[16]; const float* bdv = (const float*)d_in[17];
  const float* wo  = (const float*)d_in[18]; const float* bo  = (const float*)d_in[19];
  const float* wgg = (const float*)d_in[20]; const float* bgg = (const float*)d_in[21];
  const float* gg_g = (const float*)d_in[22]; const float* gg_b = (const float*)d_in[23];
  const float* wdg = (const float*)d_in[24]; const float* bdg = (const float*)d_in[25];
  const float* dg_g = (const float*)d_in[26]; const float* dg_b = (const float*)d_in[27];
  float* out = (float*)d_out;

  char* ws = (char*)d_ws;
  size_t off = 0;
  auto alloc = [&](size_t bytes) -> char* {
    char* p = ws + off;
    off = (off + bytes + 255) & ~(size_t)255;
    return p;
  };
  const size_t f16SD = (size_t)S * D * sizeof(_Float16);   // 4 MB
  const size_t f32SD = (size_t)S * D * sizeof(float);      // 8 MB
  const size_t f16DD = (size_t)D * D * sizeof(_Float16);   // 0.5 MB

  _Float16* wgq16 = (_Float16*)alloc(f16DD);
  _Float16* wgk16 = (_Float16*)alloc(f16DD);
  _Float16* wgv16 = (_Float16*)alloc(f16DD);
  _Float16* wdq16 = (_Float16*)alloc(f16DD);
  _Float16* wdk16 = (_Float16*)alloc(f16DD);
  _Float16* wdv16 = (_Float16*)alloc(f16DD);
  _Float16* wo16  = (_Float16*)alloc(f16DD);
  _Float16* wgg16 = (_Float16*)alloc(2 * f16DD);
  _Float16* wdg16 = (_Float16*)alloc(2 * f16DD);
  _Float16* embg16 = (_Float16*)alloc(f16SD);
  _Float16* embd16 = (_Float16*)alloc(f16SD);
  _Float16* n16    = (_Float16*)alloc(f16SD);
  float*    tmp    = (float*)alloc(f32SD);
  _Float16* qh_g = (_Float16*)alloc(f16SD);
  _Float16* kh_g = (_Float16*)alloc(f16SD);
  _Float16* vh_g = (_Float16*)alloc(f16SD);
  _Float16* qh_d = (_Float16*)alloc(f16SD);
  _Float16* kh_d = (_Float16*)alloc(f16SD);
  _Float16* vh_d = (_Float16*)alloc(f16SD);
  float* ctx_g = (float*)alloc(f32SD);
  float* ctx_d = (float*)alloc(f32SD);
  _Float16* ctx16 = (_Float16*)alloc(f16SD);
  _Float16* cat16 = (_Float16*)alloc(2 * f16SD);
  float* gate_g = (float*)alloc(f32SD);
  float* gate_d = (float*)alloc(f32SD);
  float* proj_g = (float*)alloc(f32SD);
  float* proj_d = (float*)alloc(f32SD);
  (void)ws_size; (void)n_in; (void)in_sizes; (void)out_size;

  auto cvt = [&](const float* src, _Float16* dst, int n) {
    cvt_f16_kernel<<<(n + 255) / 256, 256, 0, stream>>>(src, dst, n);
  };
  auto gemm = [&](const _Float16* A, const _Float16* B, const float* bias,
                  float* C, int M, int N, int K) {
    gemm_wmma_kernel<<<dim3(M / 128, N / 64), 256, 0, stream>>>(A, B, bias, C, M, N, K);
  };
  auto split = [&](const float* X, _Float16* o, float scale, int do_norm) {
    split_heads_kernel<<<(S * H) / 8, 256, 0, stream>>>(X, o, S, scale, do_norm);
  };

  // --- weight / embed conversion to f16 ---
  cvt(wgq, wgq16, D * D);  cvt(wgk, wgk16, D * D);  cvt(wgv, wgv16, D * D);
  cvt(wdq, wdq16, D * D);  cvt(wdk, wdk16, D * D);  cvt(wdv, wdv16, D * D);
  cvt(wo,  wo16,  D * D);  cvt(wgg, wgg16, 2 * D * D);  cvt(wdg, wdg16, 2 * D * D);
  cvt(gene, embg16, S * D);
  cvt(drug, embd16, S * D);

  const float qscale = 0.125f;   // DH^-0.5 = 1/8

  // --- gene projections ---
  ln_f16_kernel<<<S, 256, 0, stream>>>(gene, lng_g, lng_b, n16);
  gemm(n16, wgq16, bgq, tmp, S, D, D);    split(tmp, qh_g, qscale, 1);
  gemm(n16, wgk16, bgk, tmp, S, D, D);    split(tmp, kh_g, 1.0f, 1);
  gemm(embg16, wgv16, bgv, tmp, S, D, D); split(tmp, vh_g, 1.0f, 0);

  // --- drug projections ---
  ln_f16_kernel<<<S, 256, 0, stream>>>(drug, lnd_g, lnd_b, n16);
  gemm(n16, wdq16, bdq, tmp, S, D, D);    split(tmp, qh_d, qscale, 1);
  gemm(n16, wdk16, bdk, tmp, S, D, D);    split(tmp, kh_d, 1.0f, 1);
  gemm(embd16, wdv16, bdv, tmp, S, D, D); split(tmp, vh_d, 1.0f, 0);

  // --- cross attention ---
  attn_wmma_kernel<<<dim3(S / 128, H), 256, 0, stream>>>(qh_g, kh_d, vh_d, ctx_g, S, S);
  attn_wmma_kernel<<<dim3(S / 128, H), 256, 0, stream>>>(qh_d, kh_g, vh_g, ctx_d, S, S);

  // --- gene gate + output ---
  cvt(ctx_g, ctx16, S * D);
  gemm(ctx16, wo16, bo, proj_g, S, D, D);
  concat_f16_kernel<<<(S * 2 * D) / 256, 256, 0, stream>>>(ctx_g, gene, cat16);
  gemm(cat16, wgg16, bgg, gate_g, S, D, 2 * D);
  gate_ln_sigmoid_kernel<<<S, 256, 0, stream>>>(gate_g, gg_g, gg_b);
  final_out_kernel<<<(S * D) / 256, 256, 0, stream>>>(gene, gate_g, proj_g, out, S * D);

  // --- drug gate + output ---
  cvt(ctx_d, ctx16, S * D);
  gemm(ctx16, wo16, bo, proj_d, S, D, D);
  concat_f16_kernel<<<(S * 2 * D) / 256, 256, 0, stream>>>(ctx_d, drug, cat16);
  gemm(cat16, wdg16, bdg, gate_d, S, D, 2 * D);
  gate_ln_sigmoid_kernel<<<S, 256, 0, stream>>>(gate_d, dg_g, dg_b);
  final_out_kernel<<<(S * D) / 256, 256, 0, stream>>>(drug, gate_d, proj_d,
                                                      out + (size_t)S * D, S * D);
}
